// AutoregressiveMultiGNNv1_62551903699298
// MI455X (gfx1250) — matrix operand order repaired and checked by reading, not compiled, not run on hardware
//
#include <hip/hip_runtime.h>
#include <hip/hip_fp16.h>

typedef __attribute__((ext_vector_type(16))) _Float16 v16h;
typedef __attribute__((ext_vector_type(8)))  _Float16 v8h;
typedef __attribute__((ext_vector_type(8)))  float    v8f;

// ---------------------------------------------------------------------------
// Problem constants (match reference)
// ---------------------------------------------------------------------------
#define NNODES 8000
#define NEDGES 128000
#define NCONF  2
#define NR     (NNODES*NCONF)       // 16000 node rows (enc)
#define ERL    ((long)NEDGES*NCONF) // 256000 edge rows (enc)

// ---------------------------------------------------------------------------
// Segmented operand descriptors (shared host/device POD)
//   A-row r is the concatenation of up to 4 segments.
//   value = base[k], base = (sel&&!sel[r] ? p1 : p0) + idx(r)*ld
// ---------------------------------------------------------------------------
struct SegS {
    const float* p0[4]; const float* p1[4];
    const int*   idx[4]; const int*  sel[4];
    int ld[4]; int w[4]; int n;
};
struct SegV {
    const float* p0[3]; const float* p1[3];
    const int*   idx[3]; const int*  sel[3];
    int ld[3]; int w[3]; int n;
};

// ---------------------------------------------------------------------------
// Stage A operand: gather + select + concat + f32->f16, K padded to mult of 32.
// k varies fastest across threads -> segment branch is near-uniform per wave.
// ---------------------------------------------------------------------------
__global__ void pack_a_kernel(SegS A, int rows, int K, int Kpad,
                              _Float16* __restrict__ out)
{
    long t = blockIdx.x * (long)blockDim.x + threadIdx.x;
    if (t >= (long)rows * Kpad) return;
    int r = (int)(t / Kpad), k = (int)(t % Kpad);
    float v = 0.f;
    int c0 = 0;
#pragma unroll
    for (int s = 0; s < 4; s++) {
        if (s < A.n) {
            int c1 = c0 + A.w[s];
            if (k >= c0 && k < c1) {
                int rr = A.idx[s] ? A.idx[s][r] : r;
                const float* p = A.p0[s];
                if (A.sel[s] && !A.sel[s][r]) p = A.p1[s];
                v = p[(long)rr * A.ld[s] + (k - c0)];
            }
            c0 = c1;
        }
    }
    out[t] = (_Float16)v;
}

// Stage B operand: weights (N x K row-major f32) -> f16 with Kpad stride.
__global__ void pack_w_kernel(const float* __restrict__ W, int N, int K, int Kpad,
                              _Float16* __restrict__ out)
{
    long t = blockIdx.x * (long)blockDim.x + threadIdx.x;
    if (t >= (long)N * Kpad) return;
    int n = (int)(t / Kpad), k = (int)(t % Kpad);
    out[t] = (_Float16)(k < K ? W[(long)n * K + k] : 0.f);
}

// ---------------------------------------------------------------------------
// C[M x N] = act( Apacked[M x Kpad] @ Bpacked[N x Kpad]^T + bias )
// One wave per 16(M) x 64(N) macro-tile: 4 accumulators share one A fragment,
// amortizing A loads 4x (2 A loads + 8 B loads per 4 WMMAs).
// A frag layout (contiguous k order, same as B): lane-half kh reads 16 contiguous K at k0+16*kh
// B frag layout: lane-half kh reads 16 contiguous K at k0+16*kh
// act: 0 none, 1 silu, 2 sigmoid
// ---------------------------------------------------------------------------
__global__ void gemm_wmma_f16(const _Float16* __restrict__ Ap,
                              const _Float16* __restrict__ Bp,
                              int M, int N, int Kpad,
                              const float* __restrict__ bias,
                              float* __restrict__ C, int ldc, int act, int nt4)
{
    int wave = threadIdx.x >> 5;
    int lane = threadIdx.x & 31;
    int tile = blockIdx.x * (blockDim.x >> 5) + wave;
    int mt = tile / nt4, nb = tile % nt4;
    int m0 = mt * 16, nbase = nb * 64;
    if (m0 >= M) return;

    int l15 = lane & 15, kh = lane >> 4;
    int mrow = m0 + l15; if (mrow >= M) mrow = M - 1;     // clamp; store masked
    const _Float16* ar = Ap + (long)mrow * Kpad;

    const _Float16* br[4];
    v8f acc[4];
#pragma unroll
    for (int j = 0; j < 4; j++) {
        int nrow = nbase + j * 16 + l15;
        int nclp = (nrow < N) ? nrow : (N - 1);
        br[j] = Bp + (long)nclp * Kpad;
        float binit = (bias && nrow < N) ? bias[nrow] : 0.f;
#pragma unroll
        for (int i = 0; i < 8; i++) acc[j][i] = binit;
    }

    for (int k0 = 0; k0 < Kpad; k0 += 32) {
        union { v16h v; v8h h[2]; } a;
        a.h[0] = *(const v8h*)(ar + k0 + kh * 16);
        a.h[1] = *(const v8h*)(ar + k0 + kh * 16 + 8);
#pragma unroll
        for (int j = 0; j < 4; j++) {
            v16h b = *(const v16h*)(br[j] + k0 + kh * 16);
            acc[j] = __builtin_amdgcn_wmma_f32_16x16x32_f16(false, a.v, false, b,
                                                            (short)0, acc[j],
                                                            false, false);
        }
    }

#pragma unroll
    for (int j = 0; j < 4; j++) {
        int n = nbase + j * 16 + l15;
        if (n >= N) continue;
#pragma unroll
        for (int r = 0; r < 8; r++) {
            int m = m0 + r + 8 * kh;
            if (m < M) {
                float x = acc[j][r];
                if (act == 1)      x = x / (1.f + __expf(-x));   // silu
                else if (act == 2) x = 1.f / (1.f + __expf(-x)); // sigmoid
                C[(long)m * ldc + n] = x;
            }
        }
    }
}

// ---------------------------------------------------------------------------
// vh = concatV[rows x Kv x 3] @ wh^T  (wh is H x Kv), plus vn = ||vh|| (clipped)
// ---------------------------------------------------------------------------
__global__ void vh_norm_kernel(SegV V, int rows, int Kv, int H,
                               const float* __restrict__ wh,
                               float* __restrict__ vh, float* __restrict__ vn)
{
    long t = blockIdx.x * (long)blockDim.x + threadIdx.x;
    if (t >= (long)rows * H) return;
    int row = (int)(t / H);
    int hi  = (int)(t % H);

    const float* base[3];
    int cum[4]; cum[0] = 0;
    for (int s = 0; s < V.n; s++) {
        cum[s+1] = cum[s] + V.w[s];
        int r = V.idx[s] ? V.idx[s][row] : row;
        const float* src = V.p0[s];
        if (V.sel[s] && !V.sel[s][row]) src = V.p1[s];
        base[s] = src + (long)r * V.ld[s];
    }
    const float* wr = wh + (long)hi * Kv;
    float ax = 0.f, ay = 0.f, az = 0.f;
    for (int s = 0; s < V.n; s++) {
        const float* b = base[s];
        int w0 = cum[s];
        for (int j = 0; j < V.w[s]; j++) {
            float w = wr[w0 + j];
            ax += w * b[j*3+0]; ay += w * b[j*3+1]; az += w * b[j*3+2];
        }
    }
    long o = (long)row * H + hi;
    vh[o*3+0] = ax; vh[o*3+1] = ay; vh[o*3+2] = az;
    vn[o] = sqrtf(fmaxf(ax*ax + ay*ay + az*az, 1e-8f));
}

// vout[row,o,:] = gate[row,o] * sum_j wv[o,j] * vh[row,j,:]  (gate pre-sigmoided)
__global__ void vout_gate_kernel(const float* __restrict__ vh,
                                 const float* __restrict__ wv,
                                 const float* __restrict__ gate,
                                 float* __restrict__ vout,
                                 int rows, int H, int VO)
{
    long t = blockIdx.x * (long)blockDim.x + threadIdx.x;
    if (t >= (long)rows * VO) return;
    long r = t / VO; int o = (int)(t % VO);
    const float* wr = wv + (long)o * H;
    const float* vr = vh + r * (long)H * 3;
    float ax = 0.f, ay = 0.f, az = 0.f;
    for (int j = 0; j < H; j++) {
        float w = wr[j];
        ax += w * vr[j*3]; ay += w * vr[j*3+1]; az += w * vr[j*3+2];
    }
    float g = gate[t];
    vout[t*3+0] = ax*g; vout[t*3+1] = ay*g; vout[t*3+2] = az*g;
}

// ---------------------------------------------------------------------------
// GVP-style layer norm: scalar LN + vector RMS over channels, one block/row
// ---------------------------------------------------------------------------
__global__ void layer_norm_kernel(const float* __restrict__ s_in,
                                  const float* __restrict__ v_in,
                                  const float* __restrict__ gg,
                                  const float* __restrict__ bb,
                                  float* __restrict__ s_out,
                                  float* __restrict__ v_out,
                                  int ds, int dv)
{
    int row = blockIdx.x;
    int tid = threadIdx.x;
    __shared__ float red[128];
    __shared__ float bc;

    const float* sr = s_in + (long)row * ds;
    float a = 0.f;
    for (int i = tid; i < ds; i += 128) a += sr[i];
    red[tid] = a; __syncthreads();
    for (int o = 64; o > 0; o >>= 1) { if (tid < o) red[tid] += red[tid+o]; __syncthreads(); }
    if (tid == 0) bc = red[0] / ds;
    __syncthreads();
    float mu = bc; __syncthreads();

    a = 0.f;
    for (int i = tid; i < ds; i += 128) { float d = sr[i] - mu; a += d * d; }
    red[tid] = a; __syncthreads();
    for (int o = 64; o > 0; o >>= 1) { if (tid < o) red[tid] += red[tid+o]; __syncthreads(); }
    if (tid == 0) bc = red[0] / ds;
    __syncthreads();
    float inv = rsqrtf(bc + 1e-5f); __syncthreads();

    float* so = s_out + (long)row * ds;
    for (int i = tid; i < ds; i += 128) so[i] = (sr[i] - mu) * inv * gg[i] + bb[i];

    const float* vr = v_in + (long)row * dv * 3;
    a = 0.f;
    for (int j = tid; j < dv; j += 128) {
        float x = vr[j*3], y = vr[j*3+1], z = vr[j*3+2];
        a += fmaxf(x*x + y*y + z*z, 1e-8f);
    }
    red[tid] = a; __syncthreads();
    for (int o = 64; o > 0; o >>= 1) { if (tid < o) red[tid] += red[tid+o]; __syncthreads(); }
    if (tid == 0) bc = sqrtf(red[0] / dv);
    __syncthreads();
    float invv = 1.f / bc;
    float* vo = v_out + (long)row * dv * 3;
    for (int i = tid; i < dv * 3; i += 128) vo[i] = vr[i] * invv;
}

// ---------------------------------------------------------------------------
// misc elementwise / graph kernels
// ---------------------------------------------------------------------------
__global__ void build_idx_kernel(const int* __restrict__ src, const int* __restrict__ dst,
                                 int* srcRow, int* dstRow, int* fwd, int E)
{
    int e = blockIdx.x * blockDim.x + threadIdx.x;
    if (e >= E) return;
    srcRow[2*e] = src[e]*2; srcRow[2*e+1] = src[e]*2 + 1;
    dstRow[2*e] = dst[e]*2; dstRow[2*e+1] = dst[e]*2 + 1;
    fwd[e] = (src[e] < dst[e]) ? 1 : 0;
}

__global__ void deg_kernel(const int* __restrict__ dst, float* deg, int E)
{
    int e = blockIdx.x * blockDim.x + threadIdx.x;
    if (e >= E) return;
    atomicAdd(&deg[dst[e]], 1.0f);
}

__global__ void scatter_add_kernel(const float* __restrict__ x,
                                   const int* __restrict__ rowidx,
                                   float* out, long rows, int ld)
{
    long t = blockIdx.x * (long)blockDim.x + threadIdx.x;
    if (t >= rows * ld) return;
    long r = t / ld; int f = (int)(t % ld);
    atomicAdd(&out[(long)rowidx[r] * ld + f], x[t]);
}

__global__ void residual_div_add_kernel(float* s, const float* __restrict__ dh,
                                        const float* __restrict__ deg,
                                        long rows, int ld, int shift)
{
    long t = blockIdx.x * (long)blockDim.x + threadIdx.x;
    if (t >= rows * ld) return;
    long r = t / ld;
    s[t] += dh[t] / fmaxf(deg[r >> shift], 1.f);
}

__global__ void add_kernel(float* a, const float* __restrict__ b, long n)
{
    long t = blockIdx.x * (long)blockDim.x + threadIdx.x;
    if (t < n) a[t] += b[t];
}

__global__ void silu_kernel(float* x, long n)
{
    long t = blockIdx.x * (long)blockDim.x + threadIdx.x;
    if (t < n) { float v = x[t]; x[t] = v / (1.f + __expf(-v)); }
}

__global__ void pool_conf_kernel(const float* __restrict__ x,
                                 const float* __restrict__ mask,
                                 float* out, int n, int ld)
{
    long t = blockIdx.x * (long)blockDim.x + threadIdx.x;
    if (t >= (long)n * ld) return;
    int i = (int)(t / ld); int f = (int)(t % ld);
    float m0 = mask[i*2], m1 = mask[i*2+1];
    out[t] = (x[(long)(i*2) * ld + f] * m0 + x[(long)(i*2+1) * ld + f] * m1) / (m0 + m1);
}

__global__ void pool_edge_kernel(const float* __restrict__ x,
                                 const float* __restrict__ mask,
                                 const int* __restrict__ src,
                                 float* out, int E, int ld)
{
    long t = blockIdx.x * (long)blockDim.x + threadIdx.x;
    if (t >= (long)E * ld) return;
    int e = (int)(t / ld); int f = (int)(t % ld);
    int nd = src[e];
    float m0 = mask[nd*2], m1 = mask[nd*2+1];
    out[t] = (x[(long)(e*2) * ld + f] * m0 + x[(long)(e*2+1) * ld + f] * m1) / (m0 + m1);
}

__global__ void hecat_kernel(const float* __restrict__ he32,
                             const float* __restrict__ w_s,
                             const int* __restrict__ seq,
                             const int* __restrict__ src,
                             const int* __restrict__ dst,
                             float* out, int E)
{
    long t = blockIdx.x * (long)blockDim.x + threadIdx.x;
    if (t >= (long)E * 36) return;
    int e = (int)(t / 36); int f = (int)(t % 36);
    if (f < 32) out[t] = he32[(long)e*32 + f];
    else {
        int fw = src[e] < dst[e];
        out[t] = fw ? w_s[seq[src[e]]*4 + (f - 32)] : 0.f;
    }
}

// ---------------------------------------------------------------------------
// Host-side weight bookkeeping (JAX tree-flatten order: dict keys sorted)
// ---------------------------------------------------------------------------
struct LinW  { const float* b; const float* w; };
struct LNW   { const float* b; const float* g; };
struct GVPW  { const float* wh; LinW ws; LinW wsv; const float* wv;
               int si, vi, so, vo, h; };
struct ConvW { GVPW ff0, ff1; LNW ln0, ln1; GVPW m0, m1, m2; };

static inline void addSegS(SegS& S, const float* p0, const float* p1,
                           const int* idx, const int* sel, int ld, int w)
{ int i = S.n; S.p0[i]=p0; S.p1[i]=p1; S.idx[i]=idx; S.sel[i]=sel; S.ld[i]=ld; S.w[i]=w; S.n=i+1; }

static inline void addSegV(SegV& V, const float* p0, const float* p1,
                           const int* idx, const int* sel, int ld, int w)
{ int i = V.n; V.p0[i]=p0; V.p1[i]=p1; V.idx[i]=idx; V.sel[i]=sel; V.ld[i]=ld; V.w[i]=w; V.n=i+1; }

static inline unsigned blks(long n) { return (unsigned)((n + 255) / 256); }
static inline int round32(int k) { return (k + 31) & ~31; }

static void launch_gemm(hipStream_t st, _Float16* pA, _Float16* pW,
                        const SegS& A, int M, int N, int K,
                        const float* W, const float* bias, float* C, int act)
{
    int Kpad = round32(K);
    pack_a_kernel<<<blks((long)M * Kpad), 256, 0, st>>>(A, M, K, Kpad, pA);
    pack_w_kernel<<<blks((long)N * Kpad), 256, 0, st>>>(W, N, K, Kpad, pW);
    int mt = (M + 15) / 16;
    int nt4 = (N + 63) / 64;                    // 4 n-tiles per wave
    long blocks = ((long)mt * nt4 + 7) / 8;     // 8 waves (256 threads) per block
    gemm_wmma_f16<<<(unsigned)blocks, 256, 0, st>>>(pA, pW, M, N, Kpad, bias, C, N, act, nt4);
}

// Full GVP: vh/vn -> s = lin([s,vn]) -> gate = sigmoid(lin(s)) -> v = (wv@vh)*gate
// -> optional silu on s.  (vector_act = None, vector_gate = True as in reference)
static void gvp_full(hipStream_t st, _Float16* pA, _Float16* pW,
                     const GVPW& W, int rows,
                     SegS S, const SegV& V,
                     float* out_s, float* out_v,
                     float* vh_buf, float* vn_buf, float* gate_buf, int act)
{
    long tv = (long)rows * W.h;
    vh_norm_kernel<<<blks(tv), 256, 0, st>>>(V, rows, W.vi, W.h, W.wh, vh_buf, vn_buf);
    addSegS(S, vn_buf, nullptr, nullptr, nullptr, W.h, W.h);
    launch_gemm(st, pA, pW, S, rows, W.so, W.si + W.h, W.ws.w, W.ws.b, out_s, 0);
    if (W.vo) {
        SegS G{}; addSegS(G, out_s, nullptr, nullptr, nullptr, W.so, W.so);
        launch_gemm(st, pA, pW, G, rows, W.vo, W.so, W.wsv.w, W.wsv.b, gate_buf, 2);
        long to = (long)rows * W.vo;
        vout_gate_kernel<<<blks(to), 256, 0, st>>>(vh_buf, W.wv, gate_buf, out_v,
                                                   rows, W.h, W.vo);
    }
    if (act == 1) {
        long n = (long)rows * W.so;
        silu_kernel<<<blks(n), 256, 0, st>>>(out_s, n);
    }
}

// ---------------------------------------------------------------------------
extern "C" void kernel_launch(void* const* d_in, const int* in_sizes, int n_in,
                              void* d_out, int out_size, void* d_ws, size_t ws_size,
                              hipStream_t stream)
{
    (void)in_sizes; (void)n_in; (void)out_size; (void)ws_size;
    const int N = NNODES, E = NEDGES;
    const int ER = (int)ERL;

    const float* node_s = (const float*)d_in[0];   // (N,2,64)
    const float* node_v = (const float*)d_in[1];   // (N,2,4,3)
    const float* edge_s = (const float*)d_in[2];   // (E,2,32)
    const float* edge_v = (const float*)d_in[3];   // (E,2,1,3)
    const float* mask   = (const float*)d_in[4];   // (N,2)
    const int*   eidx   = (const int*)d_in[5];     // (2,E)
    const int*   seq    = (const int*)d_in[6];     // (N,)
    const int* src = eidx, *dst = eidx + E;

    // ---- param walk: flatten order = sorted dict keys, lists in order ----
    int cur = 7;
    auto nxt = [&]() -> const float* { return (const float*)d_in[cur++]; };
    auto rd_gvp = [&](int si, int vi, int so, int vo) -> GVPW {
        GVPW g; g.si=si; g.vi=vi; g.so=so; g.vo=vo;
        g.h = vo ? (vi > vo ? vi : vo) : vi;
        g.wh = nxt(); g.ws.b = nxt(); g.ws.w = nxt();
        if (vo) { g.wsv.b = nxt(); g.wsv.w = nxt(); g.wv = nxt(); }
        else { g.wsv.b = nullptr; g.wsv.w = nullptr; g.wv = nullptr; }
        return g;
    };
    auto rd_ln = [&]() -> LNW { LNW l; l.b = nxt(); l.g = nxt(); return l; };
    auto rd_conv = [&](int es, int ev) -> ConvW {
        ConvW c;                                       // keys: ff, ln0, ln1, msg
        c.ff0 = rd_gvp(128, 16, 512, 32);
        c.ff1 = rd_gvp(512, 32, 128, 16);
        c.ln0 = rd_ln(); c.ln1 = rd_ln();
        c.m0 = rd_gvp(2*128 + es, 2*16 + ev, 128, 16);
        c.m1 = rd_gvp(128, 16, 128, 16);
        c.m2 = rd_gvp(128, 16, 128, 16);
        return c;
    };
    ConvW dec[3], enc[3];
    for (int i = 0; i < 3; i++) dec[i] = rd_conv(36, 1);   // 'dec' < 'enc'
    for (int i = 0; i < 3; i++) enc[i] = rd_conv(32, 1);
    GVPW w_out = rd_gvp(128, 16, 4, 0);
    const float* w_s = nxt();
    GVPW we_gvp = rd_gvp(32, 1, 32, 1);
    LNW  we_ln  = rd_ln();
    GVPW wv_gvp = rd_gvp(64, 4, 128, 16);
    LNW  wv_ln  = rd_ln();

    // ---- scratch bump allocator ----
    char* wsb = (char*)d_ws; size_t off = 0;
    auto alloc  = [&](size_t nf) -> float* {
        float* p = (float*)(wsb + off); off += ((nf*4 + 255)/256)*256; return p; };
    auto alloch = [&](size_t nh) -> _Float16* {
        _Float16* p = (_Float16*)(wsb + off); off += ((nh*2 + 255)/256)*256; return p; };
    auto alloci = [&](size_t n) -> int* {
        int* p = (int*)(wsb + off); off += ((n*4 + 255)/256)*256; return p; };

    // packed WMMA operands (reused by every GEMM; stream-ordered)
    _Float16* pA = alloch((size_t)ER * 352);   // max rows*Kpad (enc m0: 256000 x 352)
    _Float16* pW = alloch((size_t)512 * 576);  // max N*Kpad
    // node-level
    float* s_nd  = alloc((size_t)NR*128);
    float* v_nd  = alloc((size_t)NR*48);
    float* sn_nd = alloc((size_t)NR*128);
    float* vn_nd = alloc((size_t)NR*48);
    float* dh_s  = alloc((size_t)NR*128);
    float* dh_v  = alloc((size_t)NR*48);
    float* t_s4  = alloc((size_t)NR*512);
    float* t_v2  = alloc((size_t)NR*96);
    float* nvh   = alloc((size_t)NR*96);
    float* nvn   = alloc((size_t)NR*32);
    float* ngate = alloc((size_t)NR*32);
    float* sn64  = alloc((size_t)NR*64);
    float* vn4   = alloc((size_t)NR*12);
    // edge-level (enc size, reused by dec)
    float* es_e  = alloc((size_t)ER*32);
    float* ev_e  = alloc((size_t)ER*3);
    float* esn   = alloc((size_t)ER*32);
    float* evn   = alloc((size_t)ER*3);
    float* vh_e  = alloc((size_t)ER*99);
    float* vn_e  = alloc((size_t)ER*33);
    float* gat_e = alloc((size_t)ER*16);
    float* ms_a  = alloc((size_t)ER*128);
    float* ms_b  = alloc((size_t)ER*128);
    float* mv_a  = alloc((size_t)ER*48);
    float* mv_b  = alloc((size_t)ER*48);
    // pooled / decoder
    float* hv_s  = alloc((size_t)N*128);
    float* hv_v  = alloc((size_t)N*48);
    float* enc_s = alloc((size_t)N*128);
    float* enc_v = alloc((size_t)N*48);
    float* as_b  = alloc((size_t)N*128);
    float* av_b  = alloc((size_t)N*48);
    float* he32  = alloc((size_t)E*32);
    float* hev   = alloc((size_t)E*3);
    float* hecat = alloc((size_t)E*36);
    float* deg   = alloc((size_t)N);
    int* srcRow = alloci((size_t)ER);
    int* dstRow = alloci((size_t)ER);
    int* fwd    = alloci((size_t)E);

    // ---- graph prep ----
    build_idx_kernel<<<blks(E), 256, 0, stream>>>(src, dst, srcRow, dstRow, fwd, E);
    hipMemsetAsync(deg, 0, (size_t)N*4, stream);
    deg_kernel<<<blks(E), 256, 0, stream>>>(dst, deg, E);

    // ---- input embeddings ----
    layer_norm_kernel<<<NR, 128, 0, stream>>>(node_s, node_v, wv_ln.g, wv_ln.b,
                                              sn64, vn4, 64, 4);
    { SegS S{}; addSegS(S, sn64, nullptr, nullptr, nullptr, 64, 64);
      SegV V{}; addSegV(V, vn4,  nullptr, nullptr, nullptr, 12, 4);
      gvp_full(stream, pA, pW, wv_gvp, NR, S, V, s_nd, v_nd, nvh, nvn, ngate, 0); }

    layer_norm_kernel<<<ER, 128, 0, stream>>>(edge_s, edge_v, we_ln.g, we_ln.b,
                                              esn, evn, 32, 1);
    { SegS S{}; addSegS(S, esn, nullptr, nullptr, nullptr, 32, 32);
      SegV V{}; addSegV(V, evn, nullptr, nullptr, nullptr, 3, 1);
      gvp_full(stream, pA, pW, we_gvp, ER, S, V, es_e, ev_e, vh_e, vn_e, gat_e, 0); }

    // ---- encoder ----
    for (int li = 0; li < 3; li++) {
        const ConvW& L = enc[li];
        layer_norm_kernel<<<NR, 128, 0, stream>>>(s_nd, v_nd, L.ln0.g, L.ln0.b,
                                                  sn_nd, vn_nd, 128, 16);
        { SegS S{};                                     // [sn[src] | es | sn[dst]]
          addSegS(S, sn_nd, nullptr, srcRow, nullptr, 128, 128);
          addSegS(S, es_e,  nullptr, nullptr, nullptr, 32, 32);
          addSegS(S, sn_nd, nullptr, dstRow, nullptr, 128, 128);
          SegV V{};
          addSegV(V, vn_nd, nullptr, srcRow, nullptr, 48, 16);
          addSegV(V, ev_e,  nullptr, nullptr, nullptr, 3, 1);
          addSegV(V, vn_nd, nullptr, dstRow, nullptr, 48, 16);
          gvp_full(stream, pA, pW, L.m0, ER, S, V, ms_a, mv_a, vh_e, vn_e, gat_e, 1); }
        { SegS S{}; addSegS(S, ms_a, nullptr, nullptr, nullptr, 128, 128);
          SegV V{}; addSegV(V, mv_a, nullptr, nullptr, nullptr, 48, 16);
          gvp_full(stream, pA, pW, L.m1, ER, S, V, ms_b, mv_b, vh_e, vn_e, gat_e, 1); }
        { SegS S{}; addSegS(S, ms_b, nullptr, nullptr, nullptr, 128, 128);
          SegV V{}; addSegV(V, mv_b, nullptr, nullptr, nullptr, 48, 16);
          gvp_full(stream, pA, pW, L.m2, ER, S, V, ms_a, mv_a, vh_e, vn_e, gat_e, 0); }

        hipMemsetAsync(dh_s, 0, (size_t)NR*128*4, stream);
        hipMemsetAsync(dh_v, 0, (size_t)NR*48*4, stream);
        scatter_add_kernel<<<blks((long)ER*128), 256, 0, stream>>>(ms_a, dstRow, dh_s, ER, 128);
        scatter_add_kernel<<<blks((long)ER*48),  256, 0, stream>>>(mv_a, dstRow, dh_v, ER, 48);
        residual_div_add_kernel<<<blks((long)NR*128), 256, 0, stream>>>(s_nd, dh_s, deg, NR, 128, 1);
        residual_div_add_kernel<<<blks((long)NR*48),  256, 0, stream>>>(v_nd, dh_v, deg, NR, 48, 1);

        layer_norm_kernel<<<NR, 128, 0, stream>>>(s_nd, v_nd, L.ln1.g, L.ln1.b,
                                                  sn_nd, vn_nd, 128, 16);
        { SegS S{}; addSegS(S, sn_nd, nullptr, nullptr, nullptr, 128, 128);
          SegV V{}; addSegV(V, vn_nd, nullptr, nullptr, nullptr, 48, 16);
          gvp_full(stream, pA, pW, L.ff0, NR, S, V, t_s4, t_v2, nvh, nvn, ngate, 1); }
        { SegS S{}; addSegS(S, t_s4, nullptr, nullptr, nullptr, 512, 512);
          SegV V{}; addSegV(V, t_v2, nullptr, nullptr, nullptr, 96, 32);
          gvp_full(stream, pA, pW, L.ff1, NR, S, V, dh_s, dh_v, nvh, nvn, ngate, 0); }
        add_kernel<<<blks((long)NR*128), 256, 0, stream>>>(s_nd, dh_s, (long)NR*128);
        add_kernel<<<blks((long)NR*48),  256, 0, stream>>>(v_nd, dh_v, (long)NR*48);
    }

    // ---- conformer pooling ----
    pool_conf_kernel<<<blks((long)N*128), 256, 0, stream>>>(s_nd, mask, enc_s, N, 128);
    pool_conf_kernel<<<blks((long)N*48),  256, 0, stream>>>(v_nd, mask, enc_v, N, 48);
    hipMemcpyAsync(hv_s, enc_s, (size_t)N*128*4, hipMemcpyDeviceToDevice, stream);
    hipMemcpyAsync(hv_v, enc_v, (size_t)N*48*4,  hipMemcpyDeviceToDevice, stream);
    pool_edge_kernel<<<blks((long)E*32), 256, 0, stream>>>(es_e, mask, src, he32, E, 32);
    pool_edge_kernel<<<blks((long)E*3),  256, 0, stream>>>(ev_e, mask, src, hev, E, 3);
    hecat_kernel<<<blks((long)E*36), 256, 0, stream>>>(he32, w_s, seq, src, dst, hecat, E);

    // ---- decoder ----
    for (int li = 0; li < 3; li++) {
        const ConvW& L = dec[li];
        layer_norm_kernel<<<N, 128, 0, stream>>>(hv_s, hv_v, L.ln0.g, L.ln0.b,
                                                 sn_nd, vn_nd, 128, 16);   // x
        layer_norm_kernel<<<N, 128, 0, stream>>>(enc_s, enc_v, L.ln0.g, L.ln0.b,
                                                 as_b, av_b, 128, 16);     // a
        { SegS S{};                   // [where(fwd,x,a)[src] | hecat | where(fwd,x,a)[dst]]
          addSegS(S, sn_nd, as_b, src, fwd, 128, 128);
          addSegS(S, hecat, nullptr, nullptr, nullptr, 36, 36);
          addSegS(S, sn_nd, as_b, dst, fwd, 128, 128);
          SegV V{};
          addSegV(V, vn_nd, av_b, src, fwd, 48, 16);
          addSegV(V, hev,   nullptr, nullptr, nullptr, 3, 1);
          addSegV(V, vn_nd, av_b, dst, fwd, 48, 16);
          gvp_full(stream, pA, pW, L.m0, E, S, V, ms_a, mv_a, vh_e, vn_e, gat_e, 1); }
        { SegS S{}; addSegS(S, ms_a, nullptr, nullptr, nullptr, 128, 128);
          SegV V{}; addSegV(V, mv_a, nullptr, nullptr, nullptr, 48, 16);
          gvp_full(stream, pA, pW, L.m1, E, S, V, ms_b, mv_b, vh_e, vn_e, gat_e, 1); }
        { SegS S{}; addSegS(S, ms_b, nullptr, nullptr, nullptr, 128, 128);
          SegV V{}; addSegV(V, mv_b, nullptr, nullptr, nullptr, 48, 16);
          gvp_full(stream, pA, pW, L.m2, E, S, V, ms_a, mv_a, vh_e, vn_e, gat_e, 0); }

        hipMemsetAsync(dh_s, 0, (size_t)N*128*4, stream);
        hipMemsetAsync(dh_v, 0, (size_t)N*48*4, stream);
        scatter_add_kernel<<<blks((long)E*128), 256, 0, stream>>>(ms_a, dst, dh_s, E, 128);
        scatter_add_kernel<<<blks((long)E*48),  256, 0, stream>>>(mv_a, dst, dh_v, E, 48);
        residual_div_add_kernel<<<blks((long)N*128), 256, 0, stream>>>(hv_s, dh_s, deg, N, 128, 0);
        residual_div_add_kernel<<<blks((long)N*48),  256, 0, stream>>>(hv_v, dh_v, deg, N, 48, 0);

        layer_norm_kernel<<<N, 128, 0, stream>>>(hv_s, hv_v, L.ln1.g, L.ln1.b,
                                                 sn_nd, vn_nd, 128, 16);
        { SegS S{}; addSegS(S, sn_nd, nullptr, nullptr, nullptr, 128, 128);
          SegV V{}; addSegV(V, vn_nd, nullptr, nullptr, nullptr, 48, 16);
          gvp_full(stream, pA, pW, L.ff0, N, S, V, t_s4, t_v2, nvh, nvn, ngate, 1); }
        { SegS S{}; addSegS(S, t_s4, nullptr, nullptr, nullptr, 512, 512);
          SegV V{}; addSegV(V, t_v2, nullptr, nullptr, nullptr, 96, 32);
          gvp_full(stream, pA, pW, L.ff1, N, S, V, dh_s, dh_v, nvh, nvn, ngate, 0); }
        add_kernel<<<blks((long)N*128), 256, 0, stream>>>(hv_s, dh_s, (long)N*128);
        add_kernel<<<blks((long)N*48),  256, 0, stream>>>(hv_v, dh_v, (long)N*48);
    }

    // ---- output head (GVP, no vector out) ----
    { SegS S{}; addSegS(S, hv_s, nullptr, nullptr, nullptr, 128, 128);
      SegV V{}; addSegV(V, hv_v, nullptr, nullptr, nullptr, 48, 16);
      gvp_full(stream, pA, pW, w_out, N, S, V, (float*)d_out, nullptr, nvh, nvn, ngate, 0); }
}
